// SerialTransformerLayer_69114613727697
// MI455X (gfx1250) — matrix operand order repaired and checked by reading, not compiled, not run on hardware
//
#include <hip/hip_runtime.h>
#include <hip/hip_bf16.h>

#define DIM    2048
#define NHEADS 16
#define HDIM   128
#define HIDDEN 8192
#define SEQ    2048
#define BATCH  2
#define MROWS  (BATCH * SEQ)   // 4096

typedef _Float16 f16;
typedef __attribute__((ext_vector_type(16))) _Float16 v16h;
typedef __attribute__((ext_vector_type(8)))  _Float16 v8h;
typedef __attribute__((ext_vector_type(8)))  float    v8f;
typedef __attribute__((ext_vector_type(4)))  int      i4;

union AF { v16h v; v8h h[2]; _Float16 e[16]; };
union BFr { v16h v; _Float16 e[16]; };
union CF { v8f v; float e[8]; };

// ------------------------------------------------ CDNA5 async-to-LDS support
#if defined(__has_builtin)
#if __has_builtin(__builtin_amdgcn_global_load_async_to_lds_b128)
#define HAVE_ASYNC 1
#endif
#endif
#ifndef HAVE_ASYNC
#define HAVE_ASYNC 0
#endif

#if HAVE_ASYNC
typedef i4 __attribute__((address_space(1)))* as1_i4p;  // global int4*
typedef i4 __attribute__((address_space(3)))* as3_i4p;  // LDS int4*
#if __has_builtin(__builtin_amdgcn_s_wait_asynccnt)
#define WAIT_ASYNC(n) __builtin_amdgcn_s_wait_asynccnt(n)
#else
#define WAIT_ASYNC(n) asm volatile("s_wait_asynccnt " #n ::: "memory")
#endif
#else
#define WAIT_ASYNC(n) ((void)0)
#endif

// copy 16 f16 (32 B) global -> LDS, async if available
__device__ __forceinline__ void stage16(const f16* __restrict__ g, f16* l) {
#if HAVE_ASYNC
  __builtin_amdgcn_global_load_async_to_lds_b128((as1_i4p)g, (as3_i4p)l, 0, 0);
  __builtin_amdgcn_global_load_async_to_lds_b128((as1_i4p)(g + 8), (as3_i4p)(l + 8), 0, 0);
#else
  *(v16h*)l = *(const v16h*)g;
#endif
}

__device__ __forceinline__ v8f wmma_f16(v16h a, v16h b, v8f c) {
  // 8 args: (neg_a, A, neg_b, B, c_mod, C, reuse_a, reuse_b)
  return __builtin_amdgcn_wmma_f32_16x16x32_f16(false, a, false, b, (short)0, c,
                                                false, false);
}

// ---------------------------------------------------------------- cast f32->f16
__global__ void cast_f32_f16(const float* __restrict__ s, f16* __restrict__ d, int n) {
  int i = blockIdx.x * blockDim.x + threadIdx.x;
  if (i < n) d[i] = (f16)s[i];
}

// ---------------------------------------------------------------- RMSNorm -> f16
__global__ void rmsnorm_f16(const float* __restrict__ x, const float* __restrict__ w,
                            f16* __restrict__ o) {
  __shared__ float red[256];
  const int row = blockIdx.x;
  const int t = threadIdx.x;
  const float* xr = x + (size_t)row * DIM;
  float s = 0.f;
#pragma unroll
  for (int i = 0; i < DIM / 256; i++) {
    float v = xr[t + i * 256];
    s += v * v;
  }
  red[t] = s;
  __syncthreads();
  for (int st = 128; st > 0; st >>= 1) {
    if (t < st) red[t] += red[t + st];
    __syncthreads();
  }
  // weight * x * D^-0.5 / sqrt(mean(x^2)+eps)
  const float kf = rsqrtf(red[0] / (float)DIM + 1e-6f) * 0.022097086912079608f;
#pragma unroll
  for (int i = 0; i < DIM / 256; i++) {
    int idx = t + i * 256;
    o[(size_t)row * DIM + idx] = (f16)(w[idx] * xr[idx] * kf);
  }
}

// ---------------------------------------------------------------- WMMA GEMM
// C[M,N] = A[M,K] @ W[N,K]^T   (A, W row-major, K contiguous)
// W tile (64 x 64) double-buffered in LDS, shared by all 8 waves (8x reuse).
// mode 0: oh[idx] = (f16)acc ; mode 1: of[idx] = res[idx] + acc
__global__ void gemm_wmma(const f16* __restrict__ A, const f16* __restrict__ W,
                          const float* __restrict__ res, float* __restrict__ of,
                          f16* __restrict__ oh, int M, int N, int K, int mode) {
  __shared__ __align__(32) f16 Wl[2][64 * 64];
  const int t = threadIdx.x;
  const int lane = t & 31;
  const int wave = t >> 5;
  const int lh = lane >> 4;      // lane half (0/1)
  const int col = lane & 15;
  const long mtile = (long)blockIdx.y * 128 + wave * 16;
  const long ntile = (long)blockIdx.x * 64;
  const f16* arow = A + (mtile + col) * (long)K;
  const int off0 = lh * 16;
  // staging assignment: 256 threads x 32 B cover 64 rows x 64 halves
  const int crow = t >> 2, ckseg = (t & 3) * 16;
  const f16* wsrc = W + (ntile + crow) * (long)K + ckseg;
  f16* wdst0 = &Wl[0][crow * 64 + ckseg];
  f16* wdst1 = &Wl[1][crow * 64 + ckseg];

  stage16(wsrc, wdst0);
  CF acc[4] = {};
  int buf = 0;
  for (int kb = 0; kb < K; kb += 64) {
    const bool hasnext = (kb + 64) < K;
    if (hasnext) stage16(wsrc + kb + 64, buf ? wdst0 : wdst1);
    if (hasnext) { WAIT_ASYNC(2); } else { WAIT_ASYNC(0); }
    __syncthreads();
    __builtin_prefetch(arow + kb + 64, 0, 0);
    const f16* wl = &Wl[buf][0];
#pragma unroll
    for (int ks = 0; ks < 64; ks += 32) {
      AF a;
      a.h[0] = *(const v8h*)(arow + kb + ks + off0);
      a.h[1] = *(const v8h*)(arow + kb + ks + off0 + 8);
#pragma unroll
      for (int nb = 0; nb < 4; nb++) {
        v16h b = *(const v16h*)(wl + (nb * 16 + col) * 64 + ks + lh * 16);
        acc[nb].v = wmma_f16(a.v, b, acc[nb].v);
      }
    }
    __syncthreads();
    buf ^= 1;
  }
#pragma unroll
  for (int nb = 0; nb < 4; nb++) {
#pragma unroll
    for (int r = 0; r < 8; r++) {
      long mi = mtile + r + lh * 8;
      long ni = ntile + nb * 16 + col;
      long idx = mi * (long)N + ni;
      float v = acc[nb].e[r];
      if (mode == 1)
        of[idx] = res[idx] + v;
      else
        oh[idx] = (f16)v;
    }
  }
}

// ---------------------------------------------------------------- fused SwiGLU gate
// G[M,N] = silu(A@W1^T) * (A@W3^T)  (f16 out), W1/W3 tiles staged in LDS
__global__ void gemm_gate(const f16* __restrict__ A, const f16* __restrict__ W1,
                          const f16* __restrict__ W3, f16* __restrict__ G,
                          int M, int N, int K) {
  __shared__ __align__(32) f16 W1l[2][64 * 64];
  __shared__ __align__(32) f16 W3l[2][64 * 64];
  const int t = threadIdx.x;
  const int lane = t & 31;
  const int wave = t >> 5;
  const int lh = lane >> 4;
  const int col = lane & 15;
  const long mtile = (long)blockIdx.y * 128 + wave * 16;
  const long ntile = (long)blockIdx.x * 64;
  const f16* arow = A + (mtile + col) * (long)K;
  const int off0 = lh * 16;
  const int crow = t >> 2, ckseg = (t & 3) * 16;
  const long woff = (ntile + crow) * (long)K + ckseg;
  const int loff = crow * 64 + ckseg;

  stage16(W1 + woff, &W1l[0][loff]);
  stage16(W3 + woff, &W3l[0][loff]);
  CF a1[4] = {}, a3[4] = {};
  int buf = 0;
  for (int kb = 0; kb < K; kb += 64) {
    const bool hasnext = (kb + 64) < K;
    if (hasnext) {
      stage16(W1 + woff + kb + 64, &W1l[buf ^ 1][loff]);
      stage16(W3 + woff + kb + 64, &W3l[buf ^ 1][loff]);
      WAIT_ASYNC(4);
    } else {
      WAIT_ASYNC(0);
    }
    __syncthreads();
#pragma unroll
    for (int ks = 0; ks < 64; ks += 32) {
      AF a;
      a.h[0] = *(const v8h*)(arow + kb + ks + off0);
      a.h[1] = *(const v8h*)(arow + kb + ks + off0 + 8);
#pragma unroll
      for (int nb = 0; nb < 4; nb++) {
        int li = (nb * 16 + col) * 64 + ks + lh * 16;
        v16h b1 = *(const v16h*)(&W1l[buf][li]);
        v16h b3 = *(const v16h*)(&W3l[buf][li]);
        a1[nb].v = wmma_f16(a.v, b1, a1[nb].v);
        a3[nb].v = wmma_f16(a.v, b3, a3[nb].v);
      }
    }
    __syncthreads();
    buf ^= 1;
  }
#pragma unroll
  for (int nb = 0; nb < 4; nb++) {
#pragma unroll
    for (int r = 0; r < 8; r++) {
      long mi = mtile + r + lh * 8;
      long ni = ntile + nb * 16 + col;
      float u = a1[nb].e[r];
      float g = (u / (1.f + __expf(-u))) * a3[nb].e[r];
      G[mi * (long)N + ni] = (f16)g;
    }
  }
}

// ---------------------------------------------------------------- flash attention
// q:[4096,2048] f16 (heads packed), k/v:[4096,128] f16 (MQA shared),
// o:[4096,2048] f16. One wave handles (b, head, 16 query rows).
__global__ void flash_attn(const f16* __restrict__ q, const f16* __restrict__ k,
                           const f16* __restrict__ v, f16* __restrict__ o) {
  __shared__ __align__(32) f16 Vlds[32 * HDIM];
  __shared__ __align__(32) f16 Plds[16 * 32];
  const int lane = threadIdx.x;
  const int lh = lane >> 4;
  const int col = lane & 15;
  const int tile = blockIdx.x;
  const int qb = tile & (SEQ / 16 - 1);       // 7 bits
  const int hh = (tile >> 7) & (NHEADS - 1);  // 4 bits
  const int b = tile >> 11;
  const long qrow0 = (long)b * SEQ + qb * 16;
  const int off0 = lh * 16;

  // resident Q fragments: 16x128 as 4x (16x32)
  const f16* qbase = q + (qrow0 + col) * (long)DIM + hh * HDIM;
  AF qf[4];
#pragma unroll
  for (int d = 0; d < 4; d++) {
    qf[d].h[0] = *(const v8h*)(qbase + d * 32 + off0);
    qf[d].h[1] = *(const v8h*)(qbase + d * 32 + off0 + 8);
  }

  CF o8[8] = {};
  float m_[8], l_[8];
#pragma unroll
  for (int r = 0; r < 8; r++) { m_[r] = -3.0e38f; l_[r] = 0.f; }
  const float sc = 0.08838834764831845f;  // 1/sqrt(128)

  for (int kt = 0; kt < SEQ / 32; kt++) {
    const long krow0 = (long)b * SEQ + kt * 32;

    // stage V tile (32x128 f16, contiguous) into LDS
    const unsigned int* vsrc = (const unsigned int*)(v + krow0 * HDIM);
    unsigned int* vdst = (unsigned int*)Vlds;
#pragma unroll
    for (int i = 0; i < 64; i++) vdst[i * 32 + lane] = vsrc[i * 32 + lane];
    asm volatile("s_wait_dscnt 0" ::: "memory");

    // S = Q @ K^T  (16x32 scores = two 16x16 WMMA tiles, K-dim = 128)
    CF s0 = {}, s1 = {};
#pragma unroll
    for (int d = 0; d < 4; d++) {
      v16h b0 = *(const v16h*)(k + (krow0 + col) * (long)HDIM + d * 32 + lh * 16);
      v16h b1 = *(const v16h*)(k + (krow0 + 16 + col) * (long)HDIM + d * 32 + lh * 16);
      s0.v = wmma_f16(qf[d].v, b0, s0.v);
      s1.v = wmma_f16(qf[d].v, b1, s1.v);
    }

    // online softmax: row reductions across the 16-lane half-groups
    float al[8];
#pragma unroll
    for (int r = 0; r < 8; r++) {
      s0.e[r] *= sc;
      s1.e[r] *= sc;
      float tmx = fmaxf(s0.e[r], s1.e[r]);
      tmx = fmaxf(tmx, __shfl_xor(tmx, 1, 32));
      tmx = fmaxf(tmx, __shfl_xor(tmx, 2, 32));
      tmx = fmaxf(tmx, __shfl_xor(tmx, 4, 32));
      tmx = fmaxf(tmx, __shfl_xor(tmx, 8, 32));
      float mn = fmaxf(m_[r], tmx);
      al[r] = __expf(m_[r] - mn);
      m_[r] = mn;
      float p0 = __expf(s0.e[r] - mn);
      float p1 = __expf(s1.e[r] - mn);
      s0.e[r] = p0;
      s1.e[r] = p1;
      float su = p0 + p1;
      su += __shfl_xor(su, 1, 32);
      su += __shfl_xor(su, 2, 32);
      su += __shfl_xor(su, 4, 32);
      su += __shfl_xor(su, 8, 32);
      l_[r] = l_[r] * al[r] + su;
    }

    // C-layout P -> LDS -> reload in A-fragment layout (row<->lane transpose)
#pragma unroll
    for (int r = 0; r < 8; r++) {
      int row = r + lh * 8;
      Plds[row * 32 + col] = (f16)s0.e[r];
      Plds[row * 32 + 16 + col] = (f16)s1.e[r];
    }
    asm volatile("s_wait_dscnt 0" ::: "memory");
    AF pf;
    pf.h[0] = *(const v8h*)(&Plds[col * 32 + off0]);
    pf.h[1] = *(const v8h*)(&Plds[col * 32 + off0 + 8]);

    // O = O*alpha + P @ V
#pragma unroll
    for (int nb = 0; nb < 8; nb++) {
#pragma unroll
      for (int r = 0; r < 8; r++) o8[nb].e[r] *= al[r];
      BFr vb;
#pragma unroll
      for (int j = 0; j < 16; j++) vb.e[j] = Vlds[(lh * 16 + j) * HDIM + nb * 16 + col];
      o8[nb].v = wmma_f16(pf.v, vb.v, o8[nb].v);
    }
  }

#pragma unroll
  for (int r = 0; r < 8; r++) l_[r] = 1.f / l_[r];
  f16* ob = o + hh * HDIM;
#pragma unroll
  for (int nb = 0; nb < 8; nb++) {
#pragma unroll
    for (int r = 0; r < 8; r++) {
      long row = qrow0 + r + lh * 8;
      ob[row * (long)DIM + nb * 16 + col] = (f16)(o8[nb].e[r] * l_[r]);
    }
  }
}

// ---------------------------------------------------------------- launcher
extern "C" void kernel_launch(void* const* d_in, const int* in_sizes, int n_in,
                              void* d_out, int out_size, void* d_ws, size_t ws_size,
                              hipStream_t stream) {
  (void)in_sizes; (void)n_in; (void)out_size; (void)ws_size;
  const float* x   = (const float*)d_in[0];
  const float* n1w = (const float*)d_in[1];
  const float* wq  = (const float*)d_in[2];
  const float* wk  = (const float*)d_in[3];
  const float* wv  = (const float*)d_in[4];
  const float* wo  = (const float*)d_in[5];
  const float* n2w = (const float*)d_in[6];
  const float* w1  = (const float*)d_in[7];
  const float* w2  = (const float*)d_in[8];
  const float* w3  = (const float*)d_in[9];
  float* out = (float*)d_out;

  char* p = (char*)d_ws;
  auto alloc = [&](size_t bytes) -> char* {
    char* r = p;
    p += (bytes + 255) & ~(size_t)255;
    return r;
  };
  f16* h16  = (f16*)alloc((size_t)MROWS * DIM * 2);
  f16* wq16 = (f16*)alloc((size_t)DIM * DIM * 2);
  f16* wk16 = (f16*)alloc((size_t)HDIM * DIM * 2);
  f16* wv16 = (f16*)alloc((size_t)HDIM * DIM * 2);
  f16* wo16 = (f16*)alloc((size_t)DIM * DIM * 2);
  f16* w116 = (f16*)alloc((size_t)HIDDEN * DIM * 2);
  f16* w216 = (f16*)alloc((size_t)DIM * HIDDEN * 2);
  f16* w316 = (f16*)alloc((size_t)HIDDEN * DIM * 2);
  f16* q16  = (f16*)alloc((size_t)MROWS * DIM * 2);
  f16* k16  = (f16*)alloc((size_t)MROWS * HDIM * 2);
  f16* v16_ = (f16*)alloc((size_t)MROWS * HDIM * 2);
  f16* at16 = (f16*)alloc((size_t)MROWS * DIM * 2);
  float* xr1 = (float*)alloc((size_t)MROWS * DIM * 4);
  f16* g16  = (f16*)alloc((size_t)MROWS * HIDDEN * 2);

  auto castk = [&](const float* s, f16* d, int n) {
    cast_f32_f16<<<(n + 255) / 256, 256, 0, stream>>>(s, d, n);
  };
  // Pre-cast all weights to f16: 113 MB total -> resident in the 192 MB L2.
  castk(wq, wq16, DIM * DIM);
  castk(wk, wk16, HDIM * DIM);
  castk(wv, wv16, HDIM * DIM);
  castk(wo, wo16, DIM * DIM);
  castk(w1, w116, HIDDEN * DIM);
  castk(w2, w216, DIM * HIDDEN);
  castk(w3, w316, HIDDEN * DIM);

  dim3 blk(256);
  // --- attention block ---
  rmsnorm_f16<<<MROWS, 256, 0, stream>>>(x, n1w, h16);
  gemm_wmma<<<dim3(DIM / 64, MROWS / 128), blk, 0, stream>>>(
      h16, wq16, nullptr, nullptr, q16, MROWS, DIM, DIM, 0);
  gemm_wmma<<<dim3(HDIM / 64, MROWS / 128), blk, 0, stream>>>(
      h16, wk16, nullptr, nullptr, k16, MROWS, HDIM, DIM, 0);
  gemm_wmma<<<dim3(HDIM / 64, MROWS / 128), blk, 0, stream>>>(
      h16, wv16, nullptr, nullptr, v16_, MROWS, HDIM, DIM, 0);
  flash_attn<<<BATCH * NHEADS * (SEQ / 16), 32, 0, stream>>>(q16, k16, v16_, at16);
  gemm_wmma<<<dim3(DIM / 64, MROWS / 128), blk, 0, stream>>>(
      at16, wo16, x, xr1, nullptr, MROWS, DIM, DIM, 1);
  // --- FFN block ---
  rmsnorm_f16<<<MROWS, 256, 0, stream>>>(xr1, n2w, h16);
  gemm_gate<<<dim3(HIDDEN / 64, MROWS / 128), blk, 0, stream>>>(
      h16, w116, w316, g16, MROWS, HIDDEN, DIM);
  gemm_wmma<<<dim3(DIM / 64, MROWS / 128), blk, 0, stream>>>(
      g16, w216, xr1, out, nullptr, MROWS, DIM, HIDDEN, 1);
}
